// CrossViewTokenBlock_38457137169079
// MI455X (gfx1250) — matrix operand order repaired and checked by reading, not compiled, not run on hardware
//
#include <hip/hip_runtime.h>
#include <hip/hip_bf16.h>

// ---------------------------------------------------------------------------
// CrossViewTokenBlock on MI455X (gfx1250, wave32).
// GEMMs on v_wmma_f32_16x16x32_bf16, B tiles DMA'd by the Tensor Data Mover
// (tensor_load_to_lds) into double-buffered LDS; attention V tiles staged with
// global_load_async_to_lds_b128 (ASYNCcnt). LN/softmax/GELU in fp32 VALU.
// Loads are batched ahead of WMMA groups so the matrix ops issue back-to-back.
// ---------------------------------------------------------------------------

typedef unsigned short u16;
typedef unsigned long long u64;
typedef __attribute__((ext_vector_type(16))) __bf16 v16bf;
typedef __attribute__((ext_vector_type(8)))  float  v8f;
typedef __attribute__((ext_vector_type(4)))  unsigned int v4u;
typedef __attribute__((ext_vector_type(8)))  unsigned int v8u;

#define DEV __device__ __forceinline__

DEV u16 f2bf(float f) {                 // round-to-nearest-even f32 -> bf16
  unsigned u = __float_as_uint(f);
  u += 0x7FFFu + ((u >> 16) & 1u);
  return (u16)(u >> 16);
}

union Frag { v16bf v; uint4 q[2]; u16 u[16]; };
union AccF { v8f  v; float f[8]; };

DEV v8f wmma_bf16(const Frag& a, const Frag& b, v8f c) {
  return __builtin_amdgcn_wmma_f32_16x16x32_bf16(
      false, a.v, false, b.v, (short)0, c, false, false);
}

DEV float gelu_tanh(float x) {
  float t = tanhf(0.7978845608028654f * (x + 0.044715f * x * x * x));
  return 0.5f * x * (1.0f + t);
}

DEV unsigned lds_off(const void* p) {   // generic LDS ptr -> byte offset
  return (unsigned)(u64)p;
}

// Issue a 2D TDM load: tileD0 elems (contiguous, 2B) x tileD1 rows,
// row stride = strideElems, from gaddr into LDS byte offset ldsAddr.
DEV void tdm_load_2d(u64 gaddr, unsigned ldsAddr, unsigned tileD0,
                     unsigned tileD1, unsigned strideElems) {
  v4u g0;
  g0.x = 1u;                                   // count=1, user mode
  g0.y = ldsAddr;                              // bits 63:32  lds_addr
  g0.z = (unsigned)gaddr;                      // bits 95:64  global_addr lo
  g0.w = (unsigned)((gaddr >> 32) & 0x1FFFFFFu) | (2u << 30); // addr hi | type=2
  unsigned td0 = strideElems;                  // tensor_dim0 (OOB bound)
  unsigned td1 = 1u << 20;                     // tensor_dim1 (large)
  v8u g1;
  g1[0] = 1u << 16;                            // data_size = 1 -> 2 bytes
  g1[1] = (td0 & 0xFFFFu) << 16;               // tensor_dim0[15:0] @ bits 63:48
  g1[2] = (td0 >> 16) | ((td1 & 0xFFFFu) << 16);
  g1[3] = (td1 >> 16) | (tileD0 << 16);        // tile_dim0 @ bits 127:112
  g1[4] = tileD1 & 0xFFFFu;                    // tile_dim1 (tile_dim2 = 0)
  g1[5] = strideElems;                         // tensor_dim0_stride lo
  g1[6] = 0u;
  g1[7] = 0u;
  asm volatile("tensor_load_to_lds %0, %1" :: "s"(g0), "s"(g1) : "memory");
}

DEV void async_load_b128_to_lds(unsigned ldsAddr, u64 gaddr) {
  asm volatile("global_load_async_to_lds_b128 %0, %1, off"
               :: "v"(ldsAddr), "v"(gaddr) : "memory");
}
DEV void wait_asynccnt0() {
  asm volatile("s_wait_asynccnt 0x0" ::: "memory");
}

// ---------------------------------------------------------------------------
// Weight convert + transpose: WT[n*K + k] = bf16(W[k*N + n])
// ---------------------------------------------------------------------------
__global__ __launch_bounds__(256) void tconv_kernel(
    const float* __restrict__ W, u16* __restrict__ WT, int K, int N) {
  size_t idx = (size_t)blockIdx.x * 256 + threadIdx.x;
  size_t total = (size_t)K * N;
  if (idx >= total) return;
  size_t n = idx / K, k = idx - n * K;
  WT[idx] = f2bf(W[k * (size_t)N + n]);
}

// ---------------------------------------------------------------------------
// LayerNorm (fp32 in) -> bf16 out.  One 128-thread block per row.
// ---------------------------------------------------------------------------
__global__ __launch_bounds__(128) void ln_bf16_kernel(
    const float* __restrict__ X, const float* __restrict__ g,
    const float* __restrict__ b, u16* __restrict__ Y, int Dn) {
  __shared__ float red[128];
  const size_t row = blockIdx.x;
  const float* x = X + row * (size_t)Dn;
  const int tid = threadIdx.x;
  float s = 0.f;
  for (int i = tid; i < Dn; i += 128) s += x[i];
  red[tid] = s; __syncthreads();
  for (int w = 64; w > 0; w >>= 1) { if (tid < w) red[tid] += red[tid + w]; __syncthreads(); }
  float mean = red[0] / (float)Dn;
  __syncthreads();
  float v = 0.f;
  for (int i = tid; i < Dn; i += 128) { float d = x[i] - mean; v += d * d; }
  red[tid] = v; __syncthreads();
  for (int w = 64; w > 0; w >>= 1) { if (tid < w) red[tid] += red[tid + w]; __syncthreads(); }
  float rstd = rsqrtf(red[0] / (float)Dn + 1e-6f);
  u16* y = Y + row * (size_t)Dn;
  for (int i = tid; i < Dn; i += 128)
    y[i] = f2bf((x[i] - mean) * rstd * g[i] + b[i]);
}

// ---------------------------------------------------------------------------
// bf16 WMMA GEMM:  C[M,N] = A[M,K] * BT^T   (BT stored as [N,K] bf16)
// Block = 128 thr = 4 waves; wave w owns rows [m0+16w,+16), cols [n0,+64).
// B tile (64 n x 32 k, 4KB) DMA'd by TDM into double-buffered LDS (wave 0
// issues; everyone waits TENSORcnt==0 -- free for non-issuing waves -- then
// barrier), overlapped with the 4 WMMAs on the prior tile. A fragments are
// register-prefetched one k-step ahead. All 4 B frags loaded before the WMMA
// group so the matrix ops issue back-to-back.
// flags: 1 = +bias[n], 2 = GELU, 4 = fp32 out with +resid (else bf16 out).
// ---------------------------------------------------------------------------
__global__ __launch_bounds__(128) void gemm_bf16_kernel(
    const u16* __restrict__ A, const u16* __restrict__ BT,
    const float* __restrict__ bias, const float* __restrict__ resid,
    u16* __restrict__ outB, float* __restrict__ outF,
    int M, int N, int K, int flags) {
  __shared__ __align__(16) u16 sB[2][64 * 32];

  const int lane = threadIdx.x & 31;
  const int wv   = threadIdx.x >> 5;
  const int m0   = blockIdx.y * 64 + wv * 16;
  const int n0   = blockIdx.x * 64;
  const int hi   = (lane & 16) ? 1 : 0;
  const int ka   = hi ? 8 : 0;       // A-frag k sub-offset (ISA layout)
  const int kb   = hi ? 8 : 0;       // B-frag k sub-offset (ISA layout)
  const size_t arow = (size_t)(m0 + (lane & 15)) * K;
  const bool  w0 = (wv == 0);
  const unsigned ldsB[2] = { lds_off(&sB[0][0]), lds_off(&sB[1][0]) };
  const int nk = K >> 5;

  v8f acc[4];
  for (int i = 0; i < 4; ++i) acc[i] = (v8f){0.f,0.f,0.f,0.f,0.f,0.f,0.f,0.f};

  // Prologue: DMA tile 0, prefetch A frag 0.
  if (w0) tdm_load_2d((u64)(BT + (size_t)n0 * K), ldsB[0], 32u, 64u, (unsigned)K);
  Frag aCur;
  {
    const u16* pa = A + arow + ka;
    aCur.q[0] = *(const uint4*)(pa);
    aCur.q[1] = *(const uint4*)(pa + 16);
  }
  __builtin_amdgcn_s_wait_tensorcnt(0);
  __syncthreads();

  for (int kt = 0; kt < nk; ++kt) {
    // Kick DMA for next tile; it overlaps this step's WMMAs.
    if (w0 && (kt + 1 < nk))
      tdm_load_2d((u64)(BT + (size_t)n0 * K + (size_t)(kt + 1) * 32),
                  ldsB[(kt + 1) & 1], 32u, 64u, (unsigned)K);
    Frag aNext;
    if (kt + 1 < nk) {
      const u16* pa = A + arow + (kt + 1) * 32 + ka;
      aNext.q[0] = *(const uint4*)(pa);
      aNext.q[1] = *(const uint4*)(pa + 16);
    }
    // Batch all 4 B-fragment LDS loads, then issue 4 WMMAs back-to-back.
    const u16* bbuf = &sB[kt & 1][0];
    Frag bfr[4];
    #pragma unroll
    for (int nt = 0; nt < 4; ++nt) {
      const u16* pb = bbuf + (nt * 16 + (lane & 15)) * 32 + kb;
      bfr[nt].q[0] = *(const uint4*)(pb);
      bfr[nt].q[1] = *(const uint4*)(pb + 16);
    }
    #pragma unroll
    for (int nt = 0; nt < 4; ++nt)
      acc[nt] = wmma_bf16(aCur, bfr[nt], acc[nt]);

    __builtin_amdgcn_s_wait_tensorcnt(0);   // no-op for waves 1..3
    __syncthreads();
    aCur = aNext;
  }

  #pragma unroll
  for (int nt = 0; nt < 4; ++nt) {
    AccF a; a.v = acc[nt];
    int n = n0 + nt * 16 + (lane & 15);
    float bval = (flags & 1) ? bias[n] : 0.f;
    #pragma unroll
    for (int r = 0; r < 8; ++r) {
      int mm = m0 + r + (hi ? 8 : 0);
      float v = a.f[r] + bval;
      if (flags & 2) v = gelu_tanh(v);
      size_t idx = (size_t)mm * N + n;
      if (flags & 4) outF[idx] = v + resid[idx];
      else           outB[idx] = f2bf(v);
    }
  }
}

// ---------------------------------------------------------------------------
// Flash attention w/ relative-position bias.
// Grid: (T/16, H, B), block = 32 (one wave). Per wave: 16 query rows,
// stream 32-key chunks: 4 WMMAs QK^T + online softmax (LDS does the
// C-layout -> A-layout reshuffle of P) + 4 WMMAs P*V. V chunks staged with
// global_load_async_to_lds_b128, double-buffered one chunk ahead. All loads
// for a WMMA group are batched before the group.
// ---------------------------------------------------------------------------
__global__ __launch_bounds__(32) void attn_fa_kernel(
    const u16* __restrict__ Q,   // [B*T, D] bf16 (heads packed, HD=64)
    const u16* __restrict__ KV,  // [B*T, 2D] bf16; K at col 0, V at col D
    const float* __restrict__ Wr,// [H] rel-position slope per head
    u16* __restrict__ O,         // [B*T, D] bf16
    int Tn, int Dn, int KVLD) {
  __shared__ __align__(16) float s_log[16 * 32];
  __shared__ __align__(16) u16   s_p[16 * 32];
  __shared__ __align__(16) u16   s_v[2][32 * 64];
  __shared__ float s_corr[16];
  __shared__ float s_lsum[16];

  const int lane = threadIdx.x & 31;
  const int qt = blockIdx.x, h = blockIdx.y, b = blockIdx.z;
  const int q0 = qt * 16;
  const int hi = (lane & 16) ? 1 : 0;
  const int m  = lane & 15;
  const unsigned ldsV[2] = { lds_off(&s_v[0][0]), lds_off(&s_v[1][0]) };

  // Q A-fragments for both 32-wide K chunks of HD=64.
  Frag aQ0, aQ1;
  {
    const u16* qrow = Q + ((size_t)(b * Tn + q0 + m)) * Dn + h * 64;
    int ka = hi ? 16 : 0;
    aQ0.q[0] = *(const uint4*)(qrow + ka);
    aQ0.q[1] = *(const uint4*)(qrow + ka + 8);
    aQ1.q[0] = *(const uint4*)(qrow + 32 + ka);
    aQ1.q[1] = *(const uint4*)(qrow + 32 + ka + 8);
  }
  const float wrh = Wr[h];
  float mrow = -1e30f, lrow = 0.f;   // per-row softmax state (lanes 0..15)

  v8f accO[4];
  for (int i = 0; i < 4; ++i) accO[i] = (v8f){0.f,0.f,0.f,0.f,0.f,0.f,0.f,0.f};

  // Async-stage V chunk kc into s_v[kc&1]: 32 keys x 64 dims (4KB).
  auto stageV = [&](int kc) {
    const int t0 = kc * 32;
    #pragma unroll
    for (int i = 0; i < 8; ++i) {
      int idx = i * 32 + lane;                 // 0..255 chunks of 8 bf16
      int key = idx >> 3, hb = (idx & 7) * 8;
      unsigned dst = ldsV[kc & 1] + (unsigned)((key * 64 + hb) * 2);
      u64 src = (u64)(KV + ((size_t)(b * Tn + t0 + key)) * KVLD
                      + Dn + h * 64 + hb);
      async_load_b128_to_lds(dst, src);
    }
  };

  const int nkc = Tn / 32;
  stageV(0);
  wait_asynccnt0();

  for (int kc = 0; kc < nkc; ++kc) {
    const int t0 = kc * 32;
    if (kc + 1 < nkc) stageV(kc + 1);      // overlaps this chunk's compute

    // ---- logits = Q K^T : batch all 4 K fragments, then 4 WMMAs ----
    const int kbk = hi ? 16 : 0;
    Frag bK[4];                            // [nt][k-chunk]
    #pragma unroll
    for (int nt = 0; nt < 2; ++nt) {
      const u16* krow = KV + ((size_t)(b * Tn + t0 + nt * 16 + m)) * KVLD + h * 64;
      bK[nt * 2 + 0].q[0] = *(const uint4*)(krow + kbk);
      bK[nt * 2 + 0].q[1] = *(const uint4*)(krow + kbk + 8);
      bK[nt * 2 + 1].q[0] = *(const uint4*)(krow + 32 + kbk);
      bK[nt * 2 + 1].q[1] = *(const uint4*)(krow + 32 + kbk + 8);
    }
    AccF accL[2];
    #pragma unroll
    for (int nt = 0; nt < 2; ++nt) {
      accL[nt].v = (v8f){0.f,0.f,0.f,0.f,0.f,0.f,0.f,0.f};
      accL[nt].v = wmma_bf16(aQ0, bK[nt * 2 + 0], accL[nt].v);
      accL[nt].v = wmma_bf16(aQ1, bK[nt * 2 + 1], accL[nt].v);
    }
    #pragma unroll
    for (int nt = 0; nt < 2; ++nt) {
      const int tkey = t0 + nt * 16 + m;    // lane's column = key index
      #pragma unroll
      for (int r = 0; r < 8; ++r) {
        int mm = r + (hi ? 8 : 0);
        float l = accL[nt].f[r] * 0.125f + (float)((q0 + mm) - tkey) * wrh;
        s_log[mm * 32 + nt * 16 + m] = l;
      }
    }

    // ---- online softmax: lanes 0..15 each own a query row ----
    if (lane < 16) {
      float cmax = -1e30f;
      for (int j = 0; j < 32; ++j) cmax = fmaxf(cmax, s_log[lane * 32 + j]);
      float mnew = fmaxf(mrow, cmax);
      float corr = __expf(mrow - mnew);
      float ps = 0.f;
      for (int j = 0; j < 32; ++j) {
        float pv = __expf(s_log[lane * 32 + j] - mnew);
        ps += pv;
        s_p[lane * 32 + j] = f2bf(pv);
      }
      lrow = lrow * corr + ps;
      mrow = mnew;
      s_corr[lane] = corr;
    }

    // ---- rescale running output accumulators ----
    #pragma unroll
    for (int nt = 0; nt < 4; ++nt) {
      AccF a; a.v = accO[nt];
      #pragma unroll
      for (int r = 0; r < 8; ++r) a.f[r] *= s_corr[r + (hi ? 8 : 0)];
      accO[nt] = a.v;
    }

    // ---- O += P (16x32) * V (32x64): batch P + 4 V frags, then 4 WMMAs ----
    Frag aP;
    {
      const u16* pr = &s_p[m * 32];
      int ka = hi ? 16 : 0;
      aP.q[0] = *(const uint4*)(pr + ka);
      aP.q[1] = *(const uint4*)(pr + ka + 8);
    }
    const u16* vb = &s_v[kc & 1][0];
    Frag bV[4];
    #pragma unroll
    for (int nt = 0; nt < 4; ++nt) {
      int n = nt * 16 + m;
      #pragma unroll
      for (int j = 0; j < 16; ++j) bV[nt].u[j] = vb[(kbk + j) * 64 + n];
    }
    #pragma unroll
    for (int nt = 0; nt < 4; ++nt)
      accO[nt] = wmma_bf16(aP, bV[nt], accO[nt]);

    wait_asynccnt0();                      // next V chunk resident
  }

  // Normalize and store.
  if (lane < 16) s_lsum[lane] = lrow;
  #pragma unroll
  for (int nt = 0; nt < 4; ++nt) {
    AccF a; a.v = accO[nt];
    #pragma unroll
    for (int r = 0; r < 8; ++r) {
      int mm = r + (hi ? 8 : 0);
      float v = a.f[r] / s_lsum[mm];
      O[((size_t)(b * Tn + q0 + mm)) * Dn + h * 64 + nt * 16 + m] = f2bf(v);
    }
  }
}

// ---------------------------------------------------------------------------
// Host orchestration
// ---------------------------------------------------------------------------
extern "C" void kernel_launch(void* const* d_in, const int* in_sizes, int n_in,
                              void* d_out, int out_size, void* d_ws, size_t ws_size,
                              hipStream_t stream) {
  (void)in_sizes; (void)n_in; (void)out_size; (void)ws_size;
  enum { Bb = 4, Tt = 1024, Dd = 768, Hh = 12, HID = 3072 };
  const int M = Bb * Tt;                 // 4096 rows
  const size_t BTD = (size_t)M * Dd;

  const float* tokens1 = (const float*)d_in[0];
  const float* tokens2 = (const float*)d_in[1];
  const float *g_q1=(const float*)d_in[2],  *b_q1=(const float*)d_in[3];
  const float *g_kv1=(const float*)d_in[4], *b_kv1=(const float*)d_in[5];
  const float *g_q2=(const float*)d_in[6],  *b_q2=(const float*)d_in[7];
  const float *g_kv2=(const float*)d_in[8], *b_kv2=(const float*)d_in[9];
  const float *g_f1=(const float*)d_in[10], *b_f1=(const float*)d_in[11];
  const float *g_f2=(const float*)d_in[12], *b_f2=(const float*)d_in[13];
  const float *Wq12=(const float*)d_in[14], *Wkv12=(const float*)d_in[15];
  const float *Wp12=(const float*)d_in[16], *Wr12=(const float*)d_in[17];
  const float *Wq21=(const float*)d_in[18], *Wkv21=(const float*)d_in[19];
  const float *Wp21=(const float*)d_in[20], *Wr21=(const float*)d_in[21];
  const float *Wm1a=(const float*)d_in[22], *bm1a=(const float*)d_in[23];
  const float *Wm1b=(const float*)d_in[24], *bm1b=(const float*)d_in[25];
  const float *Wm2a=(const float*)d_in[26], *bm2a=(const float*)d_in[27];
  const float *Wm2b=(const float*)d_in[28], *bm2b=(const float*)d_in[29];
  float* out = (float*)d_out;

  // Workspace carve (256B aligned).
  char* p = (char*)d_ws;
  auto carve = [&](size_t bytes) -> void* {
    void* r = (void*)p; p += (bytes + 255) & ~(size_t)255; return r;
  };
  u16* WqT12  = (u16*)carve((size_t)Dd * Dd * 2);
  u16* WkvT12 = (u16*)carve((size_t)Dd * 2 * Dd * 2);
  u16* WpT12  = (u16*)carve((size_t)Dd * Dd * 2);
  u16* WqT21  = (u16*)carve((size_t)Dd * Dd * 2);
  u16* WkvT21 = (u16*)carve((size_t)Dd * 2 * Dd * 2);
  u16* WpT21  = (u16*)carve((size_t)Dd * Dd * 2);
  u16* Wm1aT  = (u16*)carve((size_t)Dd * HID * 2);
  u16* Wm1bT  = (u16*)carve((size_t)HID * Dd * 2);
  u16* Wm2aT  = (u16*)carve((size_t)Dd * HID * 2);
  u16* Wm2bT  = (u16*)carve((size_t)HID * Dd * 2);
  u16* Xa   = (u16*)carve(BTD * 2);
  u16* Xb   = (u16*)carve(BTD * 2);
  u16* Q1   = (u16*)carve(BTD * 2);
  u16* Q2   = (u16*)carve(BTD * 2);
  u16* KVa  = (u16*)carve(BTD * 4);
  u16* KVb  = (u16*)carve(BTD * 4);
  u16* Hbuf = (u16*)carve((size_t)M * HID * 2);

  auto TC = [&](const float* W, u16* WT, int K, int N) {
    size_t total = (size_t)K * N;
    tconv_kernel<<<dim3((unsigned)((total + 255) / 256)), dim3(256), 0, stream>>>(W, WT, K, N);
  };
  auto LN = [&](const float* X, const float* g, const float* b, u16* Y) {
    ln_bf16_kernel<<<dim3(M), dim3(128), 0, stream>>>(X, g, b, Y, Dd);
  };
  auto GEMM = [&](const u16* A, const u16* BTw, const float* bias,
                  const float* resid, u16* oB, float* oF, int N, int K, int flags) {
    gemm_bf16_kernel<<<dim3(N / 64, M / 64), dim3(128), 0, stream>>>(
        A, BTw, bias, resid, oB, oF, M, N, K, flags);
  };

  // Weights -> bf16 transposed.
  TC(Wq12, WqT12, Dd, Dd);   TC(Wkv12, WkvT12, Dd, 2 * Dd); TC(Wp12, WpT12, Dd, Dd);
  TC(Wq21, WqT21, Dd, Dd);   TC(Wkv21, WkvT21, Dd, 2 * Dd); TC(Wp21, WpT21, Dd, Dd);
  TC(Wm1a, Wm1aT, Dd, HID);  TC(Wm1b, Wm1bT, HID, Dd);
  TC(Wm2a, Wm2aT, Dd, HID);  TC(Wm2b, Wm2bT, HID, Dd);

  // Attention 1: q1 = LN(t1), kv2 = LN(t2);  Attention 2: q2 = LN(t2), kv1 = LN(t1)
  LN(tokens1, g_q1,  b_q1,  Xa); GEMM(Xa, WqT12,  nullptr, nullptr, Q1,  nullptr, Dd,     Dd, 0);
  LN(tokens2, g_kv1, b_kv1, Xa); GEMM(Xa, WkvT12, nullptr, nullptr, KVa, nullptr, 2 * Dd, Dd, 0);
  LN(tokens2, g_q2,  b_q2,  Xa); GEMM(Xa, WqT21,  nullptr, nullptr, Q2,  nullptr, Dd,     Dd, 0);
  LN(tokens1, g_kv2, b_kv2, Xa); GEMM(Xa, WkvT21, nullptr, nullptr, KVb, nullptr, 2 * Dd, Dd, 0);

  attn_fa_kernel<<<dim3(Tt / 16, Hh, Bb), dim3(32), 0, stream>>>(Q1, KVa, Wr12, Xa, Tt, Dd, 2 * Dd);
  attn_fa_kernel<<<dim3(Tt / 16, Hh, Bb), dim3(32), 0, stream>>>(Q2, KVb, Wr21, Xb, Tt, Dd, 2 * Dd);

  // Output projections + residual (fp32 into d_out halves).
  GEMM(Xa, WpT12, nullptr, tokens1, nullptr, out,       Dd, Dd, 4);
  GEMM(Xb, WpT21, nullptr, tokens2, nullptr, out + BTD, Dd, Dd, 4);

  // MLP 1: t1 += W2 * gelu(W1 * LN(t1) + b1) + b2
  LN(out, g_f1, b_f1, Xa);
  GEMM(Xa,   Wm1aT, bm1a, nullptr, Hbuf,  nullptr, HID, Dd,  1 | 2);
  GEMM(Hbuf, Wm1bT, bm1b, out,     nullptr, out,   Dd,  HID, 1 | 4);

  // MLP 2
  LN(out + BTD, g_f2, b_f2, Xa);
  GEMM(Xa,   Wm2aT, bm2a, nullptr, Hbuf,    nullptr, HID, Dd,  1 | 2);
  GEMM(Hbuf, Wm2bT, bm2b, out + BTD, nullptr, out + BTD, Dd, HID, 1 | 4);
}
